// ESN_55722905698307
// MI455X (gfx1250) — matrix operand order repaired and checked by reading, not compiled, not run on hardware
//
#include <hip/hip_runtime.h>
#include <hip/hip_bf16.h>
#include <math.h>

// ---- problem constants (match reference) ----
#define T_STEPS 16384
#define NIN     64
#define NOUT    64
#define RES     2048

// ---- kernel geometry ----
#define NWG     64            // persistent workgroups (grid-sync participants)
#define ROWS    32            // reservoir rows per WG  (NWG*ROWS == RES)
#define NTHREADS 256          // 8 wave32 per WG
#define NWAVE   8
#define KSLICE  (RES / NWAVE) // 256 k-columns per wave
#define NKT     (KSLICE / 32) // 8 k-tiles of 32 per wave

typedef __attribute__((ext_vector_type(16))) __bf16 v16bf;
typedef __attribute__((ext_vector_type(8)))  float  v8f;

// ---------------------------------------------------------------------------
// Init kernel: zero output (accumulated via atomics), state vector, counter.
// Runs every launch so graph replay is deterministic.
// ---------------------------------------------------------------------------
__global__ void esn_init_kernel(float* __restrict__ out,
                                unsigned* __restrict__ ctr,
                                __bf16* __restrict__ stateG) {
    const size_t gid    = (size_t)blockIdx.x * blockDim.x + threadIdx.x;
    const size_t stride = (size_t)gridDim.x * blockDim.x;
    const size_t n      = (size_t)T_STEPS * NOUT;
    for (size_t i = gid; i < n; i += stride) out[i] = 0.0f;
    if (gid < RES) stateG[gid] = (__bf16)0.0f;
    if (gid == 0)  *ctr = 0u;
}

// ---------------------------------------------------------------------------
// Persistent cooperative ESN kernel.
//   - W_res slice lives in VGPRs (bf16 WMMA A-fragments) for all 16384 steps.
//   - Per step: broadcast state (bf16, 4KB) from L2 -> LDS, 16 WMMAs/wave,
//     cross-wave reduce via ds_add_f32, tanh, publish slice, grid sync.
// ---------------------------------------------------------------------------
__global__ __launch_bounds__(NTHREADS)
void esn_persistent_kernel(const float* __restrict__ X,
                           const float* __restrict__ W_in,
                           const float* __restrict__ W_res,
                           const float* __restrict__ W_out,
                           float* __restrict__ out,
                           unsigned* __restrict__ ctr,
                           __bf16* __restrict__ stateG) {
    const int tid   = threadIdx.x;
    const int lane  = tid & 31;
    const int wave  = tid >> 5;
    const int mrow  = lane & 15;   // M within 16-row tile
    const int khalf = lane >> 4;   // 0: K-low half, 1: K-high half (A layout)
    const int gRow0 = blockIdx.x * ROWS;

    // ---------------- LDS ----------------
    __shared__ __align__(32) __bf16 sState[RES];     // 4 KB  broadcast state
    __shared__ float sWin[ROWS * NIN];               // 8 KB  W_in slice  [r][i]
    __shared__ float sWout[NOUT * ROWS];             // 8 KB  W_out slice [o][r]
    __shared__ float sX[NIN];
    __shared__ float yAcc[ROWS];                     // cross-wave WMMA reduction
    __shared__ float nsF32[ROWS];                    // new state (fp32)

    // ---------------- one-time weight preload into VGPRs ----------------
    // A-fragment k order used here (contiguous, B uses the same) for 16x32 bf16 A:
    //   lane<16 : M=lane,    elements e=0..15 -> K=k0+e
    //   lane>=16: M=lane-16, elements e=0..15 -> K=k0+16+e
    v16bf A[2 * NKT];
#pragma unroll
    for (int rt = 0; rt < 2; ++rt) {
        const float* wrow = W_res + (size_t)(gRow0 + rt * 16 + mrow) * RES;
#pragma unroll
        for (int j = 0; j < NKT; ++j) {
            const int k0 = wave * KSLICE + j * 32;
            v16bf a;
#pragma unroll
            for (int p = 0; p < 8; ++p) {            // consecutive K pairs
                const int off = 2 * p + 16 * khalf;
                const float f0 = wrow[k0 + off];
                const float f1 = wrow[k0 + off + 1];
                a[2 * p]     = (__bf16)f0;
                a[2 * p + 1] = (__bf16)f1;
            }
            A[rt * NKT + j] = a;
        }
    }

    // W_in / W_out slices -> LDS (once)
    for (int i = tid; i < ROWS * NIN; i += NTHREADS)
        sWin[i] = W_in[(size_t)(gRow0 + i / NIN) * NIN + (i % NIN)];
    for (int i = tid; i < NOUT * ROWS; i += NTHREADS)
        sWout[i] = W_out[(size_t)(i / ROWS) * RES + gRow0 + (i % ROWS)];
    __syncthreads();

    const float inv_n = 1.0f / (float)RES;

    for (int t = 0; t < T_STEPS; ++t) {
        // Acquire: make all WGs' state stores from step t-1 visible here.
        __builtin_amdgcn_fence(__ATOMIC_ACQUIRE, "agent");

        // Broadcast state (bf16, 4 KB) -> LDS: 256 threads x 16 B.
        ((uint4*)sState)[tid] = ((const uint4*)stateG)[tid];
        if (tid < NIN)  sX[tid]   = X[(size_t)t * NIN + tid];
        if (tid < ROWS) yAcc[tid] = 0.0f;
        __syncthreads();

        // ---- W_res @ state via WMMA: B = state tile broadcast to all 16 cols.
        // B layout (32x16 bf16): lanes 0-15 elements = K 0..15, lanes 16-31 = K 16..31.
        v8f d0 = {};
        v8f d1 = {};
#pragma unroll
        for (int j = 0; j < NKT; ++j) {
            const int kb = wave * KSLICE + j * 32 + khalf * 16;   // 32B aligned
            const v16bf b = *(const v16bf*)(sState + kb);
            d0 = __builtin_amdgcn_wmma_f32_16x16x32_bf16(
                     false, A[j],       false, b, (short)0, d0, false, false);
            d1 = __builtin_amdgcn_wmma_f32_16x16x32_bf16(
                     false, A[NKT + j], false, b, (short)0, d1, false, false);
        }
        // D layout: VGPR e -> row M=e (lanes 0-15) / M=e+8 (lanes 16-31);
        // columns identical (B broadcast), so lanes 0 & 16 carry the answer.
        if (lane == 0 || lane == 16) {
            const int base = khalf * 8;
#pragma unroll
            for (int e = 0; e < 8; ++e) {
                atomicAdd(&yAcc[base + e],      d0[e]);   // rows 0..15 of slice
                atomicAdd(&yAcc[16 + base + e], d1[e]);   // rows 16..31 of slice
            }
        }
        __syncthreads();

        // ---- input projection + tanh + publish new state slice (wave 0) ----
        if (tid < ROWS) {
            float u = yAcc[tid];
            const float* wi = sWin + tid * NIN;
#pragma unroll 16
            for (int i = 0; i < NIN; ++i) u = fmaf(wi[i], sX[i], u);
            const float v = tanhf(u);
            nsF32[tid] = v;
            stateG[gRow0 + tid] = (__bf16)v;              // bf16 publish
        }
        __syncthreads();

        // ---- grid barrier arrive (release covers wave0's state stores) ----
        const unsigned target = (unsigned)(t + 1) * NWG;
        if (tid == 0) {
            __builtin_amdgcn_fence(__ATOMIC_RELEASE, "agent");
            __hip_atomic_fetch_add(ctr, 1u, __ATOMIC_RELAXED,
                                   __HIP_MEMORY_SCOPE_AGENT);
        }

        // ---- output projection overlaps the spin ----
        if (tid < NOUT) {
            float acc = 0.0f;
            const float* wo = sWout + tid * ROWS;
#pragma unroll 8
            for (int j = 0; j < ROWS; ++j) acc = fmaf(wo[j], nsF32[j], acc);
            atomicAdd(out + (size_t)t * NOUT + tid, acc * inv_n);
        }

        // ---- grid barrier wait ----
        if (tid == 0) {
            while (__hip_atomic_load(ctr, __ATOMIC_RELAXED,
                                     __HIP_MEMORY_SCOPE_AGENT) < target)
                __builtin_amdgcn_s_sleep(1);
        }
        __syncthreads();   // gate: nobody reloads state until sync passes
    }
}

// ---------------------------------------------------------------------------
// Launch wrapper.
// Inputs (setup_inputs order): X [T*64], W_in [2048*64], W_res [2048*2048],
// W_out [64*2048], all fp32. Output: [T*64] fp32.
// Workspace: [0..4): arrival counter, [256..256+4096): bf16 state vector.
// ---------------------------------------------------------------------------
extern "C" void kernel_launch(void* const* d_in, const int* in_sizes, int n_in,
                              void* d_out, int out_size, void* d_ws, size_t ws_size,
                              hipStream_t stream) {
    const float* X     = (const float*)d_in[0];
    const float* W_in  = (const float*)d_in[1];
    const float* W_res = (const float*)d_in[2];
    const float* W_out = (const float*)d_in[3];
    float*       out   = (float*)d_out;

    unsigned* ctr    = (unsigned*)d_ws;
    __bf16*   stateG = (__bf16*)((char*)d_ws + 256);

    esn_init_kernel<<<256, 256, 0, stream>>>(out, ctr, stateG);
    esn_persistent_kernel<<<NWG, NTHREADS, 0, stream>>>(
        X, W_in, W_res, W_out, out, ctr, stateG);
}
